// Nonlocal_326417514945
// MI455X (gfx1250) — compile-verified, matrix operand order checked by reading, not run on hardware
//
#include <hip/hip_runtime.h>
#include <hip/hip_bf16.h>

// ---------------------------------------------------------------------------
// Non-local attention block on gfx1250 (MI455X).
// f16 WMMA (v_wmma_f32_16x16x32_f16) for all GEMM-shaped math, with all
// fragments assembled by 16-byte vector loads (2x global_load_b128 each):
//   - activations stored f16 pixel-major [B,N,C]: A-fragment halves are two
//     contiguous 8-half runs per lane (ISA 7.12.2 16-bit A layout)
//   - conv weights pre-packed fragment-ready [tap][ks][nt][lane][16]
//   - g projection stored channel-major [B,C,N] so PV B-fragments are
//     contiguous in the key dimension
//   - P tile staged through LDS as f16, read back as 2x ds_load_b128
// ---------------------------------------------------------------------------

typedef __attribute__((ext_vector_type(16))) _Float16 v16h;
typedef __attribute__((ext_vector_type(8)))  _Float16 v8h;
typedef __attribute__((ext_vector_type(8)))  float    v8f;

union VH  { v16h v; v8h h8[2]; _Float16 e[16]; };
union VF  { v8f  v; float f[8]; };
union HV8 { v8h  v; _Float16 e[8]; };

#define NB   4
#define NC   64
#define NH   64
#define NW   64
#define NPIX 4096
#define EPSB 1e-5f

// half-index -> K for 16-bit A/B fragments (ISA 7.12.2)
__device__ __forceinline__ int kidx(int h, int g) {
    return ((h & 8) << 1) + (g << 3) + (h & 7);
}

__device__ __forceinline__ v8f wmma_f16(v16h a, v16h b, v8f c) {
    return __builtin_amdgcn_wmma_f32_16x16x32_f16(false, a, false, b,
                                                  (short)0, c, false, false);
}

// load A/B fragment from a row of >=K0+32 contiguous halves (two b128 loads)
__device__ __forceinline__ void load_frag_row(VH& f, const _Float16* row,
                                              int K0, int g) {
    f.h8[0] = *(const v8h*)(row + K0 + 8 * g);
    f.h8[1] = *(const v8h*)(row + K0 + 16 + 8 * g);
}

// ------------------------- layout repack kernels ---------------------------

// x f32 NCHW -> f16 pixel-major [B, NPIX, C]
__global__ void pack_x_kernel(const float* __restrict__ x,
                              _Float16* __restrict__ xh) {
    int i = blockIdx.x * 256 + threadIdx.x;          // (b*NPIX+p)*NC + c
    if (i >= NB * NPIX * NC) return;
    int c = i & 63, p = (i >> 6) & 4095, b = i >> 18;
    xh[i] = (_Float16)x[((size_t)(b * NC + c)) * NPIX + p];
}

// conv weight OIHW f32 -> fragment-ready f16:
// dst[(((tap*2+ks)*4+nt)*32 + lane)*16 + h] = w[oc=nt*16+lane%16][ic=ks*32+kidx(h,lane/16)][tap]
__global__ void repack_w_kernel(const float* __restrict__ src,
                                _Float16* __restrict__ dst) {
    int i = blockIdx.x * 256 + threadIdx.x;
    if (i >= 9 * 2 * 4 * 32 * 16) return;
    int h    = i & 15;
    int lane = (i >> 4) & 31;
    int nt   = (i >> 9) & 3;
    int ks   = (i >> 11) & 1;
    int tap  = i >> 12;
    int ic = ks * 32 + kidx(h, lane >> 4);
    int oc = nt * 16 + (lane & 15);
    dst[i] = (_Float16)src[oc * (NC * 9) + ic * 9 + tap];
}

// ------------------------- 3x3 conv via WMMA -------------------------------
// One wave per 16-pixel tile (single image row). 9 taps x 2 k-steps x
// 4 n-tiles = 72 v_wmma. All fragment loads are 16-byte vector loads.
__global__ __launch_bounds__(32)
void conv3x3_wmma_kernel(const _Float16* __restrict__ xin,   // [B,NPIX,C] f16
                         const _Float16* __restrict__ wt,    // fragment-ready f16
                         const float* __restrict__ bias,     // [Cout] or null
                         float* __restrict__ out_chw,        // f32 [B,C,NPIX] or null
                         _Float16* __restrict__ out_pm,      // f16 [B,NPIX,C] or null
                         _Float16* __restrict__ out_cm) {    // f16 [B,C,NPIX] or null
    int wave = blockIdx.x;                // B * 256 waves
    int b  = wave >> 8;
    int q0 = (wave & 255) << 4;
    int yy = q0 >> 6;
    int x0 = q0 & 63;
    int lane = threadIdx.x;
    int g  = lane >> 4;
    int ln = lane & 15;

    VF acc[4];
    #pragma unroll
    for (int t = 0; t < 4; ++t)
        #pragma unroll
        for (int r = 0; r < 8; ++r) acc[t].f[r] = 0.f;

    for (int dy = -1; dy <= 1; ++dy) {
        int yq = yy + dy;
        if (yq < 0 || yq >= NH) continue;
        for (int dx = -1; dx <= 1; ++dx) {
            int tap = (dy + 1) * 3 + (dx + 1);
            int xm = x0 + ln + dx;                     // this lane's A-row column
            bool valid = (xm >= 0) && (xm < NW);
            const _Float16* arow =
                xin + ((size_t)(b * NPIX + yq * NW + (valid ? xm : 0))) * NC;
            #pragma unroll
            for (int ks = 0; ks < 2; ++ks) {
                VH a;
                load_frag_row(a, arow, ks * 32, g);
                if (!valid) {
                    #pragma unroll
                    for (int h = 0; h < 16; ++h) a.e[h] = (_Float16)0.f;
                }
                const _Float16* wfrag =
                    wt + ((size_t)((tap * 2 + ks) * 4) * 32 + lane) * 16;
                #pragma unroll
                for (int t = 0; t < 4; ++t) {
                    VH bb;
                    bb.v = *(const v16h*)(wfrag + (size_t)t * (32 * 16));
                    acc[t].v = wmma_f16(a.v, bb.v, acc[t].v);
                }
            }
        }
    }

    #pragma unroll
    for (int t = 0; t < 4; ++t) {
        int ch = t * 16 + ln;
        float bv = bias ? bias[ch] : 0.f;
        if (out_chw) {                       // contiguous pixels: 1 v8f store
            VF ov;
            #pragma unroll
            for (int r = 0; r < 8; ++r) ov.f[r] = acc[t].f[r] + bv;
            *(v8f*)(out_chw + ((size_t)(b * NC + ch)) * NPIX + q0 + 8 * g) = ov.v;
        }
        if (out_cm) {                        // contiguous pixels: 1 v8h store
            HV8 oh;
            #pragma unroll
            for (int r = 0; r < 8; ++r) oh.e[r] = (_Float16)(acc[t].f[r] + bv);
            *(v8h*)(out_cm + ((size_t)(b * NC + ch)) * NPIX + q0 + 8 * g) = oh.v;
        }
        if (out_pm) {
            #pragma unroll
            for (int r = 0; r < 8; ++r) {
                int pix = q0 + r + 8 * g;
                out_pm[((size_t)(b * NPIX + pix)) * NC + ch] =
                    (_Float16)(acc[t].f[r] + bv);
            }
        }
    }
}

// ------------------------- batchnorm ---------------------------------------

__global__ __launch_bounds__(256)
void bn_stats_kernel(const float* __restrict__ raw,  // f32 [B,C,NPIX]
                     float* __restrict__ stats) {    // [0:64)=mean [64:128)=var
    __shared__ float ss[256], ss2[256];
    int c = blockIdx.x, t = threadIdx.x;
    float s = 0.f, s2 = 0.f;
    for (int j = t; j < NB * NPIX; j += 256) {
        int b = j >> 12, p = j & 4095;
        float v = raw[((size_t)(b * NC + c)) * NPIX + p];
        s += v; s2 += v * v;
    }
    ss[t] = s; ss2[t] = s2;
    __syncthreads();
    for (int k = 128; k > 0; k >>= 1) {
        if (t < k) { ss[t] += ss[t + k]; ss2[t] += ss2[t + k]; }
        __syncthreads();
    }
    if (t == 0) {
        float mean = ss[0] * (1.f / (NB * NPIX));
        float var  = ss2[0] * (1.f / (NB * NPIX)) - mean * mean;
        stats[c] = mean; stats[NC + c] = var;
    }
}

__global__ void bn_relu_pack_kernel(const float* __restrict__ raw,
                                    const float* __restrict__ stats,
                                    const float* __restrict__ gam,
                                    const float* __restrict__ bet,
                                    _Float16* __restrict__ out_pm) {
    int i = blockIdx.x * 256 + threadIdx.x;          // NCHW linear
    if (i >= NB * NC * NPIX) return;
    int p = i & 4095, c = (i >> 12) & 63, b = i >> 18;
    float v = (raw[i] - stats[c]) * rsqrtf(stats[NC + c] + EPSB) * gam[c] + bet[c];
    v = fmaxf(v, 0.f);
    out_pm[((size_t)(b * NPIX + p)) * NC + c] = (_Float16)v;
}

__global__ void bn_relu_out_kernel(const float* __restrict__ raw,
                                   const float* __restrict__ stats,
                                   const float* __restrict__ gam,
                                   const float* __restrict__ bet,
                                   float* __restrict__ out) {
    int i = blockIdx.x * 256 + threadIdx.x;
    if (i >= NB * NC * NPIX) return;
    int c = (i >> 12) & 63;
    float v = (raw[i] - stats[c]) * rsqrtf(stats[NC + c] + EPSB) * gam[c] + bet[c];
    out[i] = fmaxf(v, 0.f);
}

// ------------------------- flash attention ---------------------------------
__global__ __launch_bounds__(32)
void attn_flash_kernel(const _Float16* __restrict__ th,   // [B,N,C] f16
                       const _Float16* __restrict__ ph,   // [B,N,C] f16
                       const _Float16* __restrict__ gT,   // [B,C,N] f16 (channel-major!)
                       _Float16* __restrict__ yh) {       // [B,N,C] f16
    __shared__ __align__(16) _Float16 p_lds[16 * 32];
    int wave = blockIdx.x;                // B*256
    int b  = wave >> 8;
    int q0 = (wave & 255) << 4;
    int lane = threadIdx.x;
    int g  = lane >> 4;
    int ln = lane & 15;

    // theta A-fragments (K=64 -> two k-steps), vector loads
    VH a0, a1;
    {
        const _Float16* trow = th + ((size_t)(b * NPIX + q0 + ln)) * NC;
        load_frag_row(a0, trow, 0, g);
        load_frag_row(a1, trow, 32, g);
    }

    float mrow[8], lrow[8];
    VF o[4];
    #pragma unroll
    for (int r = 0; r < 8; ++r) { mrow[r] = -1e30f; lrow[r] = 0.f; }
    #pragma unroll
    for (int t = 0; t < 4; ++t)
        #pragma unroll
        for (int r = 0; r < 8; ++r) o[t].f[r] = 0.f;

    for (int kb = 0; kb < 128; ++kb) {
        int k0 = kb * 32;
        // prefetch next key chunk (phi rows + gT keys)
        if (kb + 1 < 128) {
            __builtin_prefetch(ph + ((size_t)(b * NPIX + k0 + 32 + ln)) * NC, 0, 1);
            __builtin_prefetch(gT + ((size_t)(b * NC + ln)) * NPIX + k0 + 32, 0, 1);
        }

        // ---- S = theta @ phi^T over this 32-key chunk
        VF s0, s1;
        #pragma unroll
        for (int r = 0; r < 8; ++r) { s0.f[r] = 0.f; s1.f[r] = 0.f; }
        {
            const _Float16* prow0 = ph + ((size_t)(b * NPIX + k0 + ln)) * NC;
            const _Float16* prow1 = prow0 + 16 * NC;
            VH b0lo, b0hi, b1lo, b1hi;
            load_frag_row(b0lo, prow0, 0, g);
            load_frag_row(b0hi, prow0, 32, g);
            load_frag_row(b1lo, prow1, 0, g);
            load_frag_row(b1hi, prow1, 32, g);
            s0.v = wmma_f16(a0.v, b0lo.v, s0.v);
            s0.v = wmma_f16(a1.v, b0hi.v, s0.v);
            s1.v = wmma_f16(a0.v, b1lo.v, s1.v);
            s1.v = wmma_f16(a1.v, b1hi.v, s1.v);
        }

        // ---- online softmax (row stats per accumulator reg; reduce across
        //      the 16 lanes holding that row's columns)
        float scale[8];
        #pragma unroll
        for (int r = 0; r < 8; ++r) {
            float rm = fmaxf(s0.f[r], s1.f[r]);
            #pragma unroll
            for (int off = 1; off < 16; off <<= 1)
                rm = fmaxf(rm, __shfl_xor(rm, off, 32));
            float mn = fmaxf(mrow[r], rm);
            float sc = __expf(mrow[r] - mn);
            float p0 = __expf(s0.f[r] - mn);
            float p1 = __expf(s1.f[r] - mn);
            float rs = p0 + p1;
            #pragma unroll
            for (int off = 1; off < 16; off <<= 1)
                rs += __shfl_xor(rs, off, 32);
            lrow[r] = lrow[r] * sc + rs;
            mrow[r] = mn;
            scale[r] = sc;
            s0.f[r] = p0; s1.f[r] = p1;
        }
        #pragma unroll
        for (int t = 0; t < 4; ++t)
            #pragma unroll
            for (int r = 0; r < 8; ++r) o[t].f[r] *= scale[r];

        // ---- transpose P (D-layout) -> A-fragment via f16 LDS
        __syncthreads();
        #pragma unroll
        for (int r = 0; r < 8; ++r) {
            p_lds[(r + 8 * g) * 32 + ln]      = (_Float16)s0.f[r];
            p_lds[(r + 8 * g) * 32 + 16 + ln] = (_Float16)s1.f[r];
        }
        __syncthreads();
        VH pa;
        load_frag_row(pa, p_lds + ln * 32, 0, g);     // 2x ds_load_b128

        // ---- O += P @ G ; G is channel-major so key-runs are contiguous
        #pragma unroll
        for (int t = 0; t < 4; ++t) {
            const _Float16* gcol =
                gT + ((size_t)(b * NC + t * 16 + ln)) * NPIX + k0;
            VH gb;
            load_frag_row(gb, gcol, 0, g);
            o[t].v = wmma_f16(pa.v, gb.v, o[t].v);
        }
    }

    // normalize and store y (f16 pixel-major for the 1x1 conv)
    #pragma unroll
    for (int r = 0; r < 8; ++r) {
        float inv = 1.f / lrow[r];
        int pix = q0 + r + 8 * g;
        #pragma unroll
        for (int t = 0; t < 4; ++t)
            yh[((size_t)(b * NPIX + pix)) * NC + t * 16 + ln] =
                (_Float16)(o[t].f[r] * inv);
    }
}

// ------------------------- 1x1 W conv + residual ---------------------------
__global__ void wconv_res_kernel(const _Float16* __restrict__ yh,  // [B,N,C] f16
                                 const float* __restrict__ Ww,     // [O,C] f32
                                 const float* __restrict__ Wb,     // [O]
                                 const float* __restrict__ x,      // f32 NCHW
                                 _Float16* __restrict__ zh) {      // [B,N,C] f16
    int i = blockIdx.x * 256 + threadIdx.x;      // (b*NPIX+p)*NC + o
    if (i >= NB * NPIX * NC) return;
    int o = i & 63, p = (i >> 6) & 4095, b = i >> 18;
    const v8h* yv = (const v8h*)(yh + (size_t)(i - o));
    float s = Wb[o];
    #pragma unroll
    for (int j = 0; j < 8; ++j) {
        HV8 yc; yc.v = yv[j];
        #pragma unroll
        for (int e = 0; e < 8; ++e)
            s += Ww[o * NC + j * 8 + e] * (float)yc.e[e];
    }
    s += x[((size_t)(b * NC + o)) * NPIX + p];
    zh[i] = (_Float16)s;
}

// ---------------------------------------------------------------------------
extern "C" void kernel_launch(void* const* d_in, const int* in_sizes, int n_in,
                              void* d_out, int out_size, void* d_ws, size_t ws_size,
                              hipStream_t stream) {
    const float* x       = (const float*)d_in[0];
    const float* conv1_w = (const float*)d_in[1];
    const float* bn1_g   = (const float*)d_in[2];
    const float* bn1_b   = (const float*)d_in[3];
    const float* theta_w = (const float*)d_in[4];
    const float* theta_b = (const float*)d_in[5];
    const float* phi_w   = (const float*)d_in[6];
    const float* phi_b   = (const float*)d_in[7];
    const float* g_w     = (const float*)d_in[8];
    const float* g_b     = (const float*)d_in[9];
    const float* W_w     = (const float*)d_in[10];
    const float* W_b     = (const float*)d_in[11];
    const float* conv2_w = (const float*)d_in[12];
    const float* bn2_g   = (const float*)d_in[13];
    const float* bn2_b   = (const float*)d_in[14];
    float* out = (float*)d_out;

    char* ws = (char*)d_ws;
    const size_t SZ_H = (size_t)NB * NPIX * NC * sizeof(_Float16);   // 2 MB
    _Float16* xh     = (_Float16*)(ws);
    _Float16* x1h    = (_Float16*)(ws + 1 * SZ_H);
    _Float16* thetah = (_Float16*)(ws + 2 * SZ_H);
    _Float16* phih   = (_Float16*)(ws + 3 * SZ_H);
    _Float16* gTh    = (_Float16*)(ws + 4 * SZ_H);   // channel-major [B,C,N]
    _Float16* yh     = (_Float16*)(ws + 5 * SZ_H);
    _Float16* zh     = (_Float16*)(ws + 6 * SZ_H);
    float*    raw    = (float*)   (ws + 7 * SZ_H);                   // 4 MB
    char* wbase = ws + 7 * SZ_H + (size_t)NB * NC * NPIX * sizeof(float);
    const size_t SZ_W = 128 * 1024;                                  // 72 KB used
    _Float16* whc1 = (_Float16*)(wbase + 0 * SZ_W);
    _Float16* whth = (_Float16*)(wbase + 1 * SZ_W);
    _Float16* whph = (_Float16*)(wbase + 2 * SZ_W);
    _Float16* whg  = (_Float16*)(wbase + 3 * SZ_W);
    _Float16* whc2 = (_Float16*)(wbase + 4 * SZ_W);
    float*    stats = (float*)(wbase + 5 * SZ_W);

    const int NELEM = NB * NC * NPIX;               // 1,048,576
    dim3 eb(256), eg((NELEM + 255) / 256);
    dim3 wg(144);                                   // 36864 / 256
    dim3 convg(NB * 256), wv(32);                   // 1 wave per 16-pixel tile

    pack_x_kernel<<<eg, eb, 0, stream>>>(x, xh);
    repack_w_kernel<<<wg, eb, 0, stream>>>(conv1_w, whc1);
    repack_w_kernel<<<wg, eb, 0, stream>>>(theta_w, whth);
    repack_w_kernel<<<wg, eb, 0, stream>>>(phi_w,   whph);
    repack_w_kernel<<<wg, eb, 0, stream>>>(g_w,     whg);
    repack_w_kernel<<<wg, eb, 0, stream>>>(conv2_w, whc2);

    // conv1 + BN + ReLU
    conv3x3_wmma_kernel<<<convg, wv, 0, stream>>>(xh, whc1, nullptr,
                                                  raw, nullptr, nullptr);
    bn_stats_kernel<<<dim3(NC), eb, 0, stream>>>(raw, stats);
    bn_relu_pack_kernel<<<eg, eb, 0, stream>>>(raw, stats, bn1_g, bn1_b, x1h);

    // projections: theta/phi pixel-major; g channel-major for PV fragments
    conv3x3_wmma_kernel<<<convg, wv, 0, stream>>>(x1h, whth, theta_b,
                                                  nullptr, thetah, nullptr);
    conv3x3_wmma_kernel<<<convg, wv, 0, stream>>>(x1h, whph, phi_b,
                                                  nullptr, phih, nullptr);
    conv3x3_wmma_kernel<<<convg, wv, 0, stream>>>(x1h, whg, g_b,
                                                  nullptr, nullptr, gTh);

    // fused non-local attention
    attn_flash_kernel<<<convg, wv, 0, stream>>>(thetah, phih, gTh, yh);

    // 1x1 W conv + residual
    wconv_res_kernel<<<eg, eb, 0, stream>>>(yh, W_w, W_b, x, zh);

    // conv2 + BN + ReLU -> output
    conv3x3_wmma_kernel<<<convg, wv, 0, stream>>>(zh, whc2, nullptr,
                                                  raw, nullptr, nullptr);
    bn_stats_kernel<<<dim3(NC), eb, 0, stream>>>(raw, stats);
    bn_relu_out_kernel<<<eg, eb, 0, stream>>>(raw, stats, bn2_g, bn2_b, out);
}
